// GTEASTLayer_38620345926113
// MI455X (gfx1250) — matrix operand order repaired and checked by reading, not compiled, not run on hardware
//
#include <hip/hip_runtime.h>
#include <stdint.h>

#define N_SRC   100000
#define N_DSTC  50000
#define DEG     16
#define NODE_IN 128
#define EDGE_IN 64
#define HID     64
#define K_CAT   (NODE_IN + HID)   // 192

typedef __attribute__((ext_vector_type(16))) __bf16   v16bf;
typedef __attribute__((ext_vector_type(2)))  __bf16   v2bf;
typedef __attribute__((ext_vector_type(8)))  float    v8f;
typedef __attribute__((ext_vector_type(4)))  uint32_t u32x4;
typedef __attribute__((ext_vector_type(4)))  float    f32x4;

union Frag {
  uint32_t u[8];
  u32x4    q[2];
  v16bf    v;
};

// Native converts: let the backend pick v_cvt_pk_bf16_f32 / v_cvt_bf16_f32.
__device__ __forceinline__ uint32_t pack2(float lo, float hi) {
  v2bf p;
  p.x = (__bf16)lo;
  p.y = (__bf16)hi;
  return __builtin_bit_cast(uint32_t, p);
}
__device__ __forceinline__ uint16_t f2bf(float f) {
  return __builtin_bit_cast(uint16_t, (__bf16)f);
}
__device__ __forceinline__ v8f zero8() {
  v8f z;
#pragma unroll
  for (int i = 0; i < 8; ++i) z[i] = 0.f;
  return z;
}
__device__ __forceinline__ v8f wmma_bf16(const Frag& a, const Frag& b, v8f c) {
  return __builtin_amdgcn_wmma_f32_16x16x32_bf16(false, a.v, false, b.v,
                                                 (short)0, c, false, false);
}
// Pack 16 floats (K = kb..kb+7 from p0/p1, K = 16+kb..16+kb+7 from p2/p3) into an A-fragment.
__device__ __forceinline__ void packA(Frag& f, f32x4 p0, f32x4 p1, f32x4 p2, f32x4 p3) {
  f.u[0] = pack2(p0.x, p0.y); f.u[1] = pack2(p0.z, p0.w);
  f.u[2] = pack2(p1.x, p1.y); f.u[3] = pack2(p1.z, p1.w);
  f.u[4] = pack2(p2.x, p2.y); f.u[5] = pack2(p2.z, p2.w);
  f.u[6] = pack2(p3.x, p3.y); f.u[7] = pack2(p3.z, p3.w);
}

// Stage W (Kdim x 64, row-major) into LDS transposed [n][k] as bf16, two K at a time.
__device__ __forceinline__ void stageW(uint16_t* __restrict__ sW,
                                       const float* __restrict__ W,
                                       int Kdim, int tid) {
  for (int idx = tid; idx < (Kdim / 2) * HID; idx += 256) {
    const int n = idx & 63;
    const int k = (idx >> 6) * 2;
    *(uint32_t*)&sW[n * Kdim + k] = pack2(W[k * HID + n], W[(k + 1) * HID + n]);
  }
}

// ---------------- kernel 0: v = Wa @ wa, c = ba . wa ----------------
__global__ void prep_kernel(const float* __restrict__ Wa, const float* __restrict__ ba,
                            const float* __restrict__ wa, float* __restrict__ v_ws,
                            float* __restrict__ c_ws) {
  const int i = threadIdx.x;                 // 64 threads
  float s = 0.f;
#pragma unroll 8
  for (int h = 0; h < HID; ++h) s += Wa[i * HID + h] * wa[h];
  v_ws[i] = s;
  if (i == 0) {
    float c = 0.f;
    for (int h = 0; h < HID; ++h) c += ba[h] * wa[h];
    c_ws[0] = c;
  }
}

// ---------------- kernel A: fused edge GEMMs + sparsemax + aggregate ----------------
__global__ __launch_bounds__(256) void edge_kernel(
    const float* __restrict__ node_features, const float* __restrict__ edge_features,
    const int* __restrict__ src_idx,
    const float* __restrict__ We, const float* __restrict__ be,
    const float* __restrict__ Wo, const float* __restrict__ bo,
    const float* __restrict__ v_ws, const float* __restrict__ c_ws,
    float* __restrict__ h_neigh) {
  __shared__ __align__(16) uint16_t sWe[HID * EDGE_IN];   // transposed [n][k], bf16
  __shared__ __align__(16) uint16_t sWo[HID * K_CAT];     // transposed [n][k], bf16
  __shared__ __align__(16) uint16_t sEout[8][16 * HID];   // per-wave e_out tile, bf16
  __shared__ float sV[EDGE_IN];
  __shared__ float sBe[HID];
  __shared__ float sBo[HID];

  const int tid = threadIdx.x;
  stageW(sWe, We, EDGE_IN, tid);
  stageW(sWo, Wo, K_CAT, tid);
  if (tid < 64) { sV[tid] = v_ws[tid]; sBe[tid] = be[tid]; sBo[tid] = bo[tid]; }
  __syncthreads();

  const int wave = tid >> 5;
  const int lane = tid & 31;
  const int mrow = lane & 15;       // row (edge-in-group) / N column
  const int half = lane >> 4;       // K-half selector
  const int kb   = half * 8;        // A-fragment K base within a 16-K half

  const int   dst  = blockIdx.x * 8 + wave;       // grid sized so dst < N_DSTC always
  const float cc   = c_ws[0];
  const float* erow = edge_features + (size_t)(dst * DEG + mrow) * EDGE_IN;

  // ---- GEMM1: e_out = relu(Ef @ We + be); fuse attention dot a = Ef . v
  v8f accE[4];
#pragma unroll
  for (int t = 0; t < 4; ++t) accE[t] = zero8();
  float adot = 0.f;
#pragma unroll
  for (int s = 0; s < 2; ++s) {
    const f32x4 p0 = *(const f32x4*)(erow + 32 * s + kb);
    const f32x4 p1 = *(const f32x4*)(erow + 32 * s + kb + 4);
    const f32x4 p2 = *(const f32x4*)(erow + 32 * s + 16 + kb);
    const f32x4 p3 = *(const f32x4*)(erow + 32 * s + 16 + kb + 4);
    Frag a; packA(a, p0, p1, p2, p3);
    const int k0 = 32 * s + kb, k1 = 32 * s + 16 + kb;
    adot += p0.x * sV[k0] + p0.y * sV[k0 + 1] + p0.z * sV[k0 + 2] + p0.w * sV[k0 + 3];
    adot += p1.x * sV[k0 + 4] + p1.y * sV[k0 + 5] + p1.z * sV[k0 + 6] + p1.w * sV[k0 + 7];
    adot += p2.x * sV[k1] + p2.y * sV[k1 + 1] + p2.z * sV[k1 + 2] + p2.w * sV[k1 + 3];
    adot += p3.x * sV[k1 + 4] + p3.y * sV[k1 + 5] + p3.z * sV[k1 + 6] + p3.w * sV[k1 + 7];
    const int kcol = 32 * s + half * 8;
#pragma unroll
    for (int t = 0; t < 4; ++t) {
      const int n = t * 16 + mrow;
      Frag b;
      const u32x4* wp = (const u32x4*)&sWe[n * EDGE_IN + kcol];
      b.q[0] = wp[0]; b.q[1] = wp[2];
      accE[t] = wmma_bf16(a, b, accE[t]);
    }
  }
  // bias + relu, stage e_out in LDS as bf16 [M][n]
#pragma unroll
  for (int t = 0; t < 4; ++t) {
    const int n = t * 16 + mrow;
    const float bias = sBe[n];
#pragma unroll
    for (int r = 0; r < 8; ++r) {
      float v = accE[t][r] + bias;
      v = v > 0.f ? v : 0.f;
      sEout[wave][(half * 8 + r) * HID + n] = f2bf(v);
    }
  }
  float aval = adot + __shfl_xor(adot, 16, 32) + cc;     // full 64-K dot
  aval = aval > 0.f ? aval : 0.01f * aval;               // leaky_relu
  __syncthreads();

  // ---- GEMM2: m = relu([h_src | e_out] @ Wo + bo), K = 192
  v8f accM[4];
#pragma unroll
  for (int t = 0; t < 4; ++t) accM[t] = zero8();
  const float* srow = node_features + (size_t)src_idx[dst * DEG + mrow] * NODE_IN;
#pragma unroll
  for (int s = 0; s < 6; ++s) {
    Frag a;
    if (s < 4) {
      const f32x4 p0 = *(const f32x4*)(srow + 32 * s + kb);
      const f32x4 p1 = *(const f32x4*)(srow + 32 * s + kb + 4);
      const f32x4 p2 = *(const f32x4*)(srow + 32 * s + 16 + kb);
      const f32x4 p3 = *(const f32x4*)(srow + 32 * s + 16 + kb + 4);
      packA(a, p0, p1, p2, p3);
    } else {
      const int off = mrow * HID + 32 * (s - 4);
      a.q[0] = *(const u32x4*)&sEout[wave][off + kb];
      a.q[1] = *(const u32x4*)&sEout[wave][off + 16 + kb];
    }
    const int kcol = 32 * s + half * 8;
#pragma unroll
    for (int t = 0; t < 4; ++t) {
      const int n = t * 16 + mrow;
      Frag b;
      const u32x4* wp = (const u32x4*)&sWo[n * K_CAT + kcol];
      b.q[0] = wp[0]; b.q[1] = wp[2];
      accM[t] = wmma_bf16(a, b, accM[t]);
    }
  }

  // ---- sparsemax over the 16 edges of this dst (sort-free support-set form)
  float z = aval;
  float zmax = z;
#pragma unroll
  for (int d = 1; d < 16; d <<= 1) zmax = fmaxf(zmax, __shfl_xor(zmax, d, 32));
  z -= zmax;
  float ki = 0.f, si = 0.f;
#pragma unroll
  for (int j = 0; j < 16; ++j) {
    const float zj = __shfl(z, j, 32);
    const bool ge = (zj > z) || (zj == z && j <= mrow);   // stable rank
    ki += ge ? 1.f : 0.f;
    si += ge ? zj : 0.f;
  }
  const bool insup = (1.f + ki * z) > si;
  float cnt = insup ? 1.f : 0.f;
  float zs  = insup ? z : 0.f;
#pragma unroll
  for (int d = 1; d < 16; d <<= 1) { cnt += __shfl_xor(cnt, d, 32); zs += __shfl_xor(zs, d, 32); }
  const float tau   = (zs - 1.f) / cnt;
  const float alpha = fmaxf(z - tau, 0.f);
  float alph[8];
#pragma unroll
  for (int r = 0; r < 8; ++r) alph[r] = __shfl(alpha, half * 8 + r, 32);

  // ---- h_neigh = sum_d alpha_d * relu(m_d + bo)
#pragma unroll
  for (int t = 0; t < 4; ++t) {
    const int n = t * 16 + mrow;
    const float bias = sBo[n];
    float part = 0.f;
#pragma unroll
    for (int r = 0; r < 8; ++r) {
      const float m = fmaxf(accM[t][r] + bias, 0.f);
      part += alph[r] * m;
    }
    part += __shfl_xor(part, 16, 32);
    if (half == 0) h_neigh[(size_t)dst * HID + n] = part;
  }
}

// ---------------- kernel B: out = relu([h_dst | h_neigh] @ Wn + bn) ----------------
__global__ __launch_bounds__(256) void out_kernel(
    const float* __restrict__ node_features, const float* __restrict__ h_neigh,
    const float* __restrict__ Wn, const float* __restrict__ bn,
    float* __restrict__ out) {
  __shared__ __align__(16) uint16_t sWn[HID * K_CAT];
  __shared__ float sBn[HID];
  const int tid = threadIdx.x;
  stageW(sWn, Wn, K_CAT, tid);
  if (tid < 64) sBn[tid] = bn[tid];
  __syncthreads();

  const int wave = tid >> 5, lane = tid & 31;
  const int mrow = lane & 15, half = lane >> 4, kb = half * 8;
  const int rowBase = (blockIdx.x * 8 + wave) * 16;
  const int row  = rowBase + mrow;
  const int rowc = row < N_DSTC ? row : N_DSTC - 1;

  const float* arow = node_features + (size_t)rowc * NODE_IN;
  const float* hrow = h_neigh + (size_t)rowc * HID;

  v8f acc[4];
#pragma unroll
  for (int t = 0; t < 4; ++t) acc[t] = zero8();
#pragma unroll
  for (int s = 0; s < 6; ++s) {
    const float* base = (s < 4) ? (arow + 32 * s) : (hrow + 32 * (s - 4));
    const f32x4 p0 = *(const f32x4*)(base + kb);
    const f32x4 p1 = *(const f32x4*)(base + kb + 4);
    const f32x4 p2 = *(const f32x4*)(base + 16 + kb);
    const f32x4 p3 = *(const f32x4*)(base + 16 + kb + 4);
    Frag a; packA(a, p0, p1, p2, p3);
    const int kcol = 32 * s + half * 8;
#pragma unroll
    for (int t = 0; t < 4; ++t) {
      Frag b;
      const u32x4* wp = (const u32x4*)&sWn[(t * 16 + mrow) * K_CAT + kcol];
      b.q[0] = wp[0]; b.q[1] = wp[2];
      acc[t] = wmma_bf16(a, b, acc[t]);
    }
  }
#pragma unroll
  for (int t = 0; t < 4; ++t) {
    const int n = t * 16 + mrow;
    const float bias = sBn[n];
#pragma unroll
    for (int r = 0; r < 8; ++r) {
      const int orow = rowBase + half * 8 + r;
      if (orow < N_DSTC) {
        const float v = acc[t][r] + bias;
        out[(size_t)orow * HID + n] = v > 0.f ? v : 0.f;
      }
    }
  }
}

extern "C" void kernel_launch(void* const* d_in, const int* in_sizes, int n_in,
                              void* d_out, int out_size, void* d_ws, size_t ws_size,
                              hipStream_t stream) {
  const float* node_features = (const float*)d_in[0];
  const float* edge_features = (const float*)d_in[1];
  const int*   src_idx       = (const int*)d_in[2];
  const float* We = (const float*)d_in[3];
  const float* be = (const float*)d_in[4];
  const float* Wa = (const float*)d_in[5];
  const float* ba = (const float*)d_in[6];
  const float* wa = (const float*)d_in[7];
  const float* Wo = (const float*)d_in[8];
  const float* bo = (const float*)d_in[9];
  const float* Wn = (const float*)d_in[10];
  const float* bn = (const float*)d_in[11];
  float* out = (float*)d_out;

  float* v_ws    = (float*)d_ws;        // 64 floats
  float* c_ws    = v_ws + 64;           // 1 float
  float* h_neigh = v_ws + 128;          // N_DSTC * HID floats (12.8 MB)

  prep_kernel<<<1, 64, 0, stream>>>(Wa, ba, wa, v_ws, c_ws);
  edge_kernel<<<N_DSTC / 8, 256, 0, stream>>>(node_features, edge_features, src_idx,
                                              We, be, Wo, bo, v_ws, c_ws, h_neigh);
  out_kernel<<<(N_DSTC + 127) / 128, 256, 0, stream>>>(node_features, h_neigh, Wn, bn, out);
}
